// CustomNBFNet_90692529422645
// MI455X (gfx1250) — matrix operand order repaired and checked by reading, not compiled, not run on hardware
//
#include <hip/hip_runtime.h>
#include <math.h>

#define NN 100000
#define EE 2000000
#define DD 32
#define LL 6
#define RR 50
#define BB 64
#define KK 416          // 13 * 32
#define KC 13
#define NTILES (NN/16)  // 6250, exact

typedef __attribute__((ext_vector_type(16))) __bf16 v16bf;
typedef __attribute__((ext_vector_type(8)))  __bf16 v8bf;
typedef __attribute__((ext_vector_type(8)))  float  v8f;
typedef __attribute__((__vector_size__(16))) int    vi4;   // matches async-LDS builtin pointee
typedef __attribute__((address_space(1))) vi4*       gvi4p;
typedef __attribute__((address_space(3))) vi4*       lvi4p;

// ---- float atomic max/min via ordered-int trick (valid for mixed signs) ----
__device__ inline void atomicMaxF(float* a, float v) {
  if (v >= 0.0f) atomicMax((int*)a, __float_as_int(v));
  else           atomicMin((unsigned int*)a, __float_as_uint(v));
}
__device__ inline void atomicMinF(float* a, float v) {
  if (v >= 0.0f) atomicMin((int*)a, __float_as_int(v));
  else           atomicMax((unsigned int*)a, __float_as_uint(v));
}

// ---- one-time setup kernels ----
__global__ __launch_bounds__(256) void k_zero(float* p, long n) {
  long i = (long)blockIdx.x * 256 + threadIdx.x;
  if (i < n) p[i] = 0.0f;
}

__global__ __launch_bounds__(256) void k_boundary(const int* __restrict__ h_index,
                                                  float* __restrict__ boundary) {
  int i = blockIdx.x * 256 + threadIdx.x;
  if (i >= BB * DD) return;
  int n = h_index[i / DD];
  boundary[(size_t)n * DD + (i % DD)] = 1.0f;
}

__global__ __launch_bounds__(256) void k_degree(const int* __restrict__ ei,
                                                const float* __restrict__ ew,
                                                float* __restrict__ degw,
                                                float* __restrict__ cntf) {
  int e = blockIdx.x * 256 + threadIdx.x;
  if (e >= EE) return;
  int v = ei[2 * e + 1];
  atomicAdd(degw + v, ew[e]);
  atomicAdd(cntf + v, 1.0f);
}

__global__ __launch_bounds__(256) void k_reduce_logdeg(const float* __restrict__ degw,
                                                       float* __restrict__ ssum) {
  __shared__ float sh[256];
  int i = blockIdx.x * 256 + threadIdx.x;
  sh[threadIdx.x] = (i < NN) ? logf(degw[i] + 1.0f) : 0.0f;
  __syncthreads();
  for (int s = 128; s > 0; s >>= 1) {
    if ((int)threadIdx.x < s) sh[threadIdx.x] += sh[threadIdx.x + s];
    __syncthreads();
  }
  if (threadIdx.x == 0) atomicAdd(ssum, sh[0]);
}

__global__ __launch_bounds__(256) void k_scales(const float* __restrict__ degw,
                                                const float* __restrict__ ssum,
                                                float* __restrict__ scales) {
  int n = blockIdx.x * 256 + threadIdx.x;
  if (n >= NN) return;
  float smean = ssum[0] / (float)NN;
  float sc = logf(degw[n] + 1.0f) / smean;
  scales[n * 3 + 0] = 1.0f;
  scales[n * 3 + 1] = sc;
  scales[n * 3 + 2] = 1.0f / fmaxf(sc, 0.01f);
}

// Swizzle W (L,416,32) into a WMMA B-operand layout (this code's contiguous k order, same as A), bf16:
// wb[l][c][t][lane][e]  where B elem (K = (lane/16)*16 + e + 32c, N = 16t + lane%16)
__global__ __launch_bounds__(256) void k_prepw(const float* __restrict__ W,
                                               __bf16* __restrict__ wb) {
  int i = blockIdx.x * 256 + threadIdx.x;
  const int TOT = LL * KC * 2 * 32 * 16;
  if (i >= TOT) return;
  int e = i & 15;
  int L = (i >> 4) & 31;
  int t = (i >> 9) & 1;
  int c = (i >> 10) % KC;
  int l = (i >> 10) / KC;
  int kk = c * 32 + (L >> 4) * 16 + e;
  int nn = t * 16 + (L & 15);
  wb[i] = (__bf16)W[((size_t)l * KK + kk) * DD + nn];
}

// ---- per-layer kernels ----
// Fold the boundary self-message (w=1) into the stat init; cnt >= 1 always.
__global__ __launch_bounds__(256) void k_stats_init(const float* __restrict__ bd,
                                                    float* __restrict__ sm, float* __restrict__ sq,
                                                    float* __restrict__ mx, float* __restrict__ mn) {
  long i = (long)blockIdx.x * 256 + threadIdx.x;
  if (i >= (long)NN * DD) return;
  float b = bd[i];
  sm[i] = b; sq[i] = b * b; mx[i] = b; mn[i] = b;
}

__global__ __launch_bounds__(256) void k_edges(const int* __restrict__ ei,
                                               const int* __restrict__ et,
                                               const float* __restrict__ ew,
                                               const float* __restrict__ x,
                                               const float* __restrict__ rel,
                                               float* __restrict__ sm, float* __restrict__ sq,
                                               float* __restrict__ mx, float* __restrict__ mn) {
  int e = blockIdx.x * 256 + threadIdx.x;
  if (e >= EE) return;
  int u = ei[2 * e + 0];
  int v = ei[2 * e + 1];
  int t = et[e];
  float w = ew[e];
  const float4* xr = (const float4*)(x + (size_t)u * DD);
  const float4* rr = (const float4*)(rel + (size_t)t * DD);
  __builtin_prefetch(xr, 0, 0);  // global_prefetch_b8
  size_t vb = (size_t)v * DD;
#pragma unroll
  for (int q = 0; q < 8; q++) {
    float4 xv = xr[q];
    float4 rv = rr[q];
    float m[4] = {xv.x * rv.x, xv.y * rv.y, xv.z * rv.z, xv.w * rv.w};
#pragma unroll
    for (int j = 0; j < 4; j++) {
      size_t idx = vb + q * 4 + j;
      float mw = m[j] * w;
      atomicAdd(sm + idx, mw);
      atomicAdd(sq + idx, m[j] * m[j] * w);
      atomicMaxF(mx + idx, mw);
      atomicMinF(mn + idx, mw);
    }
  }
}

// Build bf16 WMMA A-operand rows: [x(32) | ((mean,max,min,std) interleaved by d) x scales(3)]
// For a fixed d, the 12 outputs occupy contiguous columns 32+12d..32+12d+11 (8B-aligned),
// so pack them into three 8-byte stores.
__global__ __launch_bounds__(256) void k_build_xin(const float* __restrict__ x,
                                                   const float* __restrict__ sm, const float* __restrict__ sq,
                                                   const float* __restrict__ mx, const float* __restrict__ mn,
                                                   const float* __restrict__ cntf,
                                                   const float* __restrict__ scales,
                                                   __bf16* __restrict__ xin) {
  long i = (long)blockIdx.x * 256 + threadIdx.x;
  if (i >= (long)NN * DD) return;
  int n = (int)(i / DD), d = (int)(i % DD);
  float cnt  = cntf[n] + 1.0f;
  float mean = sm[i] / cnt;
  float sqm  = sq[i] / cnt;
  float sd   = sqrtf(fmaxf(sqm - mean * mean, 1e-6f));
  float f[4] = {mean, mx[i], mn[i], sd};
  float s0 = scales[n * 3 + 0], s1 = scales[n * 3 + 1], s2 = scales[n * 3 + 2];
  __bf16 tmp[12];
#pragma unroll
  for (int s = 0; s < 4; s++) {
    tmp[s * 3 + 0] = (__bf16)(f[s] * s0);
    tmp[s * 3 + 1] = (__bf16)(f[s] * s1);
    tmp[s * 3 + 2] = (__bf16)(f[s] * s2);
  }
  __bf16* row = xin + (size_t)n * KK;
  row[d] = (__bf16)x[i];
  uint2* dst = (uint2*)(row + DD + d * 12);
  const uint2* srcp = (const uint2*)tmp;
  dst[0] = srcp[0];
  dst[1] = srcp[1];
  dst[2] = srcp[2];
}

// WMMA GEMM: per wave, 16 nodes x 32 outputs; 13 k-chunks x 2 n-tiles = 26 wmma/wave.
// The layer's swizzled W (26,624 B) is staged into LDS once per block via the CDNA5
// async global->LDS path (ASYNCcnt), falling back to a plain cooperative copy.
__global__ __launch_bounds__(256) void k_gemm(const __bf16* __restrict__ xin,
                                              const __bf16* __restrict__ wbl,
                                              const float* __restrict__ bias,
                                              float* __restrict__ out) {
  __shared__ v16bf sWb[KC * 2 * 32];  // 26,624 bytes; B tiles for all 13 k-chunks x 2 n-tiles
  {
    const int CH = KC * 2 * 32 * 2;   // 1664 x 16-byte chunks
    const vi4* src = (const vi4*)wbl;
    vi4* dst = (vi4*)sWb;
#if __has_builtin(__builtin_amdgcn_global_load_async_to_lds_b128)
    for (int i = threadIdx.x; i < CH; i += 256) {
      __builtin_amdgcn_global_load_async_to_lds_b128(
          (gvi4p)(src + i), (lvi4p)(dst + i), 0, 0);
    }
#if __has_builtin(__builtin_amdgcn_s_wait_asynccnt)
    __builtin_amdgcn_s_wait_asynccnt(0);
#else
    asm volatile("s_wait_asynccnt 0" ::: "memory");
#endif
#else
    for (int i = threadIdx.x; i < CH; i += 256) dst[i] = src[i];
#endif
  }
  __syncthreads();

  int wave = threadIdx.x >> 5;
  int lane = threadIdx.x & 31;
  int tile = blockIdx.x * 8 + wave;
  if (tile >= NTILES) return;  // whole-wave exit -> EXEC all-ones inside WMMA waves
  int mlo = lane & 15;
  int hi  = lane >> 4;
  const __bf16* arow = xin + (size_t)(tile * 16 + mlo) * KK;
  float bv0 = bias[mlo];
  float bv1 = bias[16 + mlo];
  v8f acc0, acc1;
#pragma unroll
  for (int r = 0; r < 8; r++) { acc0[r] = bv0; acc1[r] = bv1; }
#pragma unroll
  for (int c = 0; c < KC; c++) {
    // A operand, this code's contiguous k order (same as B): lane L holds K = {hi*16+0..15} -> two b128 loads
    v8bf alo = *(const v8bf*)(arow + c * 32 + hi * 16);
    v8bf ahi = *(const v8bf*)(arow + c * 32 + hi * 16 + 8);
    v16bf a = __builtin_shufflevector(alo, ahi, 0,1,2,3,4,5,6,7,8,9,10,11,12,13,14,15);
    v16bf b0 = sWb[(c * 2 + 0) * 32 + lane];  // ds_load_b128 x2
    v16bf b1 = sWb[(c * 2 + 1) * 32 + lane];
    acc0 = __builtin_amdgcn_wmma_f32_16x16x32_bf16(false, a, false, b0, (short)0, acc0, false, false);
    acc1 = __builtin_amdgcn_wmma_f32_16x16x32_bf16(false, a, false, b1, (short)0, acc1, false, false);
  }
#pragma unroll
  for (int r = 0; r < 8; r++) {
    int node = tile * 16 + hi * 8 + r;  // C/D layout: VGPR r -> M = r (lanes 0-15) / 8+r (lanes 16-31)
    float v0 = acc0[r];
    float v1 = acc1[r];
    out[(size_t)node * DD + mlo]      = v0 > 0.0f ? v0 : 0.0f;
    out[(size_t)node * DD + 16 + mlo] = v1 > 0.0f ? v1 : 0.0f;
  }
}

extern "C" void kernel_launch(void* const* d_in, const int* in_sizes, int n_in,
                              void* d_out, int out_size, void* d_ws, size_t ws_size,
                              hipStream_t stream) {
  (void)in_sizes; (void)n_in; (void)out_size;
  const int*   edge_index  = (const int*)d_in[0];
  const int*   edge_type   = (const int*)d_in[1];
  const float* edge_weight = (const float*)d_in[2];
  const int*   h_index     = (const int*)d_in[3];
  const float* rel_emb     = (const float*)d_in[4];
  const float* W           = (const float*)d_in[5];
  const float* bvec        = (const float*)d_in[6];

  size_t off = 0;
  auto alloc = [&](size_t bytes) -> void* {
    void* r = (char*)d_ws + off;
    off += (bytes + 255) & ~(size_t)255;
    return r;
  };
  float* boundary = (float*)alloc(sizeof(float) * (size_t)NN * DD);
  float* degw     = (float*)alloc(sizeof(float) * NN);
  float* cntf     = (float*)alloc(sizeof(float) * NN);
  float* ssum     = (float*)alloc(sizeof(float) * 64);
  size_t zero_bytes = off;                       // zero [boundary..ssum] span (incl. pads)
  float* scales   = (float*)alloc(sizeof(float) * (size_t)NN * 3);
  float* xbuf     = (float*)alloc(sizeof(float) * (size_t)NN * DD);
  float* sm       = (float*)alloc(sizeof(float) * (size_t)NN * DD);
  float* sq       = (float*)alloc(sizeof(float) * (size_t)NN * DD);
  float* mx       = (float*)alloc(sizeof(float) * (size_t)NN * DD);
  float* mn       = (float*)alloc(sizeof(float) * (size_t)NN * DD);
  __bf16* xin     = (__bf16*)alloc(sizeof(__bf16) * (size_t)NN * KK);
  __bf16* wb      = (__bf16*)alloc(sizeof(__bf16) * (size_t)LL * KC * 2 * 32 * 16);
  if (off > ws_size) return;

  long zn = (long)(zero_bytes / 4);
  k_zero<<<dim3((unsigned)((zn + 255) / 256)), dim3(256), 0, stream>>>(boundary, zn);
  k_boundary<<<dim3((BB * DD + 255) / 256), dim3(256), 0, stream>>>(h_index, boundary);
  k_degree<<<dim3((EE + 255) / 256), dim3(256), 0, stream>>>(edge_index, edge_weight, degw, cntf);
  k_reduce_logdeg<<<dim3((NN + 255) / 256), dim3(256), 0, stream>>>(degw, ssum);
  k_scales<<<dim3((NN + 255) / 256), dim3(256), 0, stream>>>(degw, ssum, scales);
  k_prepw<<<dim3((LL * KC * 2 * 32 * 16 + 255) / 256), dim3(256), 0, stream>>>(W, wb);

  long nd = (long)NN * DD;
  unsigned ndb = (unsigned)((nd + 255) / 256);
  for (int l = 0; l < LL; l++) {
    const float* xcur = (l == 0) ? boundary : xbuf;
    float* xout = (l == LL - 1) ? (float*)d_out : xbuf;
    k_stats_init<<<dim3(ndb), dim3(256), 0, stream>>>(boundary, sm, sq, mx, mn);
    k_edges<<<dim3((EE + 255) / 256), dim3(256), 0, stream>>>(
        edge_index, edge_type, edge_weight, xcur, rel_emb + (size_t)l * RR * DD, sm, sq, mx, mn);
    k_build_xin<<<dim3(ndb), dim3(256), 0, stream>>>(xcur, sm, sq, mx, mn, cntf, scales, xin);
    k_gemm<<<dim3((NTILES + 7) / 8), dim3(256), 0, stream>>>(
        xin, wb + (size_t)l * KC * 2 * 32 * 16, bvec + l * DD, xout);
  }
}
